// LlamaEmbeddingClassifier_1606317769315
// MI455X (gfx1250) — matrix operand order repaired and checked by reading, not compiled, not run on hardware
//
#include <hip/hip_runtime.h>
#include <hip/hip_bf16.h>

typedef __bf16 bf16;
typedef __attribute__((ext_vector_type(8)))  __bf16 v8bf;
typedef __attribute__((ext_vector_type(16))) __bf16 v16bf;
typedef __attribute__((ext_vector_type(8)))  float  v8f;

#define V_SZ   128000
#define D_SZ   512
#define NH_SZ  8
#define NL_SZ  2
#define HID_SZ 1408
#define HD_SZ  64
#define HALF_SZ 32
#define B_SZ   4
#define T_SZ   508
#define NLAB_SZ 5
#define LBL_SZ 4
#define S_SZ   512
#define NSEQ   20              /* NLAB*B */
#define MROWS  (NSEQ * S_SZ)   /* 10240 */
#define EPSF   1e-5f

/* ---- CDNA5 Tensor Data Mover availability (device compile only) ---- */
#if defined(__gfx1250__) && __has_builtin(__builtin_amdgcn_tensor_load_to_lds) && \
    __has_builtin(__builtin_amdgcn_s_wait_tensorcnt)
#define HAVE_TDM 1
#else
#define HAVE_TDM 0
#endif

#if HAVE_TDM
#define SA_PAD 0   /* TDM writes the tile contiguously: row stride == 32 elems */
#else
#define SA_PAD 8
#endif

/* ---------------- helpers ---------------- */

__device__ __forceinline__ v16bf cat8(v8bf lo, v8bf hi) {
  return __builtin_shufflevector(lo, hi, 0,1,2,3,4,5,6,7,8,9,10,11,12,13,14,15);
}
__device__ __forceinline__ v8f wmma_bf16(v16bf a, v16bf b, v8f c) {
  return __builtin_amdgcn_wmma_f32_16x16x32_bf16(false, a, false, b, (short)0, c, false, false);
}
__device__ __forceinline__ v8f vzero8f() {
  v8f z;
#pragma unroll
  for (int i = 0; i < 8; i++) z[i] = 0.0f;
  return z;
}
__device__ __forceinline__ v8bf vzero8bf() {
  v8bf z;
#pragma unroll
  for (int i = 0; i < 8; i++) z[i] = (bf16)0.0f;
  return z;
}

#if HAVE_TDM
typedef __attribute__((ext_vector_type(4))) unsigned int u32x4;
typedef __attribute__((ext_vector_type(8))) int          i32x8;
typedef __attribute__((ext_vector_type(4))) int          i32x4;

/* Issue one TDM descriptor: load a (tile_rows x tile_cols) bf16 tile whose
 * rows are (row_stride) elements apart in global, into contiguous LDS at
 * lds_off. tensor_dim = (rows_avail, cols_avail) relative to the tile start,
 * so out-of-range rows/cols are zero-filled by the TDM (ISA 8.2). */
__device__ __forceinline__ void tdm_load_tile_bf16(const bf16* gsrc, unsigned lds_off,
                                                   int rows_avail, int cols_avail,
                                                   int tile_rows, int tile_cols,
                                                   long long row_stride) {
  unsigned long long ga = (unsigned long long)(size_t)gsrc;
  u32x4 g0;
  g0[0] = 1u;                                        /* count=1, user mode   */
  g0[1] = lds_off;                                   /* lds_addr (bytes)     */
  g0[2] = (unsigned)(ga & 0xffffffffu);              /* global_addr[31:0]    */
  g0[3] = (unsigned)((ga >> 32) & 0x01ffffffu) | (2u << 30); /* [56:32]|type=2 */
  unsigned d0 = (unsigned)cols_avail;                /* tensor_dim0          */
  unsigned d1 = (unsigned)rows_avail;                /* tensor_dim1          */
  unsigned st = (unsigned)row_stride;                /* tensor_dim0_stride   */
  i32x8 g1;
  g1[0] = (int)(1u << 16);                           /* data_size=2B, no pad */
  g1[1] = (int)((d0 & 0xffffu) << 16);               /* dim0[15:0]           */
  g1[2] = (int)(((d0 >> 16) & 0xffffu) | ((d1 & 0xffffu) << 16));
  g1[3] = (int)(((d1 >> 16) & 0xffffu) | ((unsigned)tile_cols << 16)); /* tile_dim0 */
  g1[4] = (int)(unsigned)tile_rows;                  /* tile_dim1, tile_dim2=0 */
  g1[5] = (int)st;                                   /* stride[31:0]         */
  g1[6] = 0;                                         /* stride[47:32], dim1_stride lo */
  g1[7] = 0;
  i32x4 z4;
  z4[0] = 0; z4[1] = 0; z4[2] = 0; z4[3] = 0;
  i32x8 z8;
#pragma unroll
  for (int i = 0; i < 8; i++) z8[i] = 0;
  /* clang-23 lane: 6-arg form (g0, g1, g2, g3, extra group, cpol) */
  __builtin_amdgcn_tensor_load_to_lds(g0, g1, z4, z4, z8, 0);
}
#endif

/* ---------------- embedding gather ---------------- */
__global__ __launch_bounds__(128)
void k_embed(const int* __restrict__ ids, const int* __restrict__ lids,
             const float* __restrict__ emb, float* __restrict__ h) {
  int row = blockIdx.x;
  int s = row & (S_SZ - 1);
  int n = row >> 9;
  int nl = n >> 2, b = n & 3;
  int tok = (s < T_SZ) ? ids[b * T_SZ + s] : lids[nl * LBL_SZ + (s - T_SZ)];
  const float4* src = (const float4*)(emb + (size_t)tok * D_SZ);
  float4* dst = (float4*)(h + (size_t)row * D_SZ);
  dst[threadIdx.x] = src[threadIdx.x];
}

/* ---------------- rmsnorm (fp32 in -> bf16 out) ---------------- */
__device__ __forceinline__ void rmsnorm_row(const float* __restrict__ xr,
                                            const float* __restrict__ w,
                                            bf16* __restrict__ orow) {
  __shared__ float red[256];
  int t = threadIdx.x;
  float2 v = ((const float2*)xr)[t];
  red[t] = v.x * v.x + v.y * v.y;
  __syncthreads();
#pragma unroll
  for (int o = 128; o > 0; o >>= 1) {
    if (t < o) red[t] += red[t + o];
    __syncthreads();
  }
  float inv = rsqrtf(red[0] * (1.0f / D_SZ) + EPSF);
  orow[2 * t]     = (bf16)(v.x * inv * w[2 * t]);
  orow[2 * t + 1] = (bf16)(v.y * inv * w[2 * t + 1]);
}

__global__ __launch_bounds__(256)
void k_rmsnorm(const float* __restrict__ x, const float* __restrict__ w,
               bf16* __restrict__ out) {
  int row = blockIdx.x;
  rmsnorm_row(x + (size_t)row * D_SZ, w, out + (size_t)row * D_SZ);
}

__global__ __launch_bounds__(256)
void k_rmsnorm_final(const float* __restrict__ h, const float* __restrict__ w,
                     bf16* __restrict__ out) {
  int n = blockIdx.x >> 2, j = blockIdx.x & 3;
  int row = n * S_SZ + (T_SZ - 1 + j);
  rmsnorm_row(h + (size_t)row * D_SZ, w, out + (size_t)blockIdx.x * D_SZ);
}

/* ---------------- generic bf16 WMMA GEMM ----------------
 * C[M][N] (+)= A_bf16[M][K] * B_f32[K][N]  (B converted in flight)
 * block tile 128x64, k-step 32, 256 threads = 8 waves, wave tile 32x32.
 * A tile staged by the Tensor Data Mover when available.
 * EPI: 0 = store f32, 1 = add to f32 (residual) */
template <int EPI>
__global__ __launch_bounds__(256)
void k_gemm(const bf16* __restrict__ A, const float* __restrict__ B,
            float* __restrict__ C, int M, int N, int K) {
  __shared__ alignas(16) bf16 sA[128][32 + SA_PAD];
  __shared__ alignas(16) bf16 sBt[64][40];
  int tid = threadIdx.x;
  int wave = tid >> 5, lane = tid & 31;
  int lr = lane & 15, lh = lane >> 4;
  int wm = (wave & 3) * 32, wn = (wave >> 2) * 32;
  int m0 = blockIdx.y * 128, n0 = blockIdx.x * 64;

  v8f acc[2][2];
#pragma unroll
  for (int i = 0; i < 2; i++)
#pragma unroll
    for (int j = 0; j < 2; j++) acc[i][j] = vzero8f();

  int brow = tid >> 3, bcol = (tid & 7) * 8; /* B: 8 floats per thread */

  for (int k0 = 0; k0 < K; k0 += 32) {
#if HAVE_TDM
    if (wave == 0) {
      tdm_load_tile_bf16(A + (size_t)m0 * K + k0,
                         (unsigned)(size_t)&sA[0][0],
                         M - m0, K - k0, 128, 32, K);
    }
#else
    {
      int arow = tid >> 1, ahalf = tid & 1;
      int gr = m0 + arow;
      v8bf a0 = vzero8bf(), a1 = vzero8bf();
      if (gr < M) {
        const bf16* ap = A + (size_t)gr * K + k0 + ahalf * 16;
        a0 = *(const v8bf*)ap;
        a1 = *(const v8bf*)(ap + 8);
      }
      *(v8bf*)&sA[tid >> 1][ahalf * 16]     = a0;
      *(v8bf*)&sA[tid >> 1][ahalf * 16 + 8] = a1;
    }
#endif
    {
      const float* bp = B + (size_t)(k0 + brow) * N + n0 + bcol;
      float4 b0 = *(const float4*)bp;
      float4 b1 = *(const float4*)(bp + 4);
      sBt[bcol + 0][brow] = (bf16)b0.x; sBt[bcol + 1][brow] = (bf16)b0.y;
      sBt[bcol + 2][brow] = (bf16)b0.z; sBt[bcol + 3][brow] = (bf16)b0.w;
      sBt[bcol + 4][brow] = (bf16)b1.x; sBt[bcol + 5][brow] = (bf16)b1.y;
      sBt[bcol + 6][brow] = (bf16)b1.z; sBt[bcol + 7][brow] = (bf16)b1.w;
      if (k0 + 32 < K)
        __builtin_prefetch(B + (size_t)(k0 + 32 + brow) * N + n0 + bcol, 0, 0);
    }
#if HAVE_TDM
    if (wave == 0) __builtin_amdgcn_s_wait_tensorcnt(0);
#endif
    __syncthreads();
    v16bf af[2], bfr[2];
#pragma unroll
    for (int i = 0; i < 2; i++) {
      const bf16* ap = &sA[wm + i * 16 + lr][lh * 8];
      af[i] = cat8(*(const v8bf*)ap, *(const v8bf*)(ap + 16));
    }
#pragma unroll
    for (int j = 0; j < 2; j++) {
      const bf16* bp = &sBt[wn + j * 16 + lr][lh * 8];
      bfr[j] = cat8(*(const v8bf*)bp, *(const v8bf*)(bp + 16));
    }
#pragma unroll
    for (int i = 0; i < 2; i++)
#pragma unroll
      for (int j = 0; j < 2; j++) acc[i][j] = wmma_bf16(af[i], bfr[j], acc[i][j]);
    __syncthreads();
  }
#pragma unroll
  for (int i = 0; i < 2; i++)
#pragma unroll
    for (int j = 0; j < 2; j++)
#pragma unroll
      for (int r = 0; r < 8; r++) {
        int row = m0 + wm + i * 16 + r + 8 * lh;
        int col = n0 + wn + j * 16 + lr;
        if (row < M) {
          float* cp = &C[(size_t)row * N + col];
          if (EPI == 0) *cp = acc[i][j][r];
          else          *cp = *cp + acc[i][j][r];
        }
      }
}

/* ---------------- fused SwiGLU GEMM: t = silu(A*B1) * (A*B3), bf16 out */
__global__ __launch_bounds__(256)
void k_swiglu(const bf16* __restrict__ A, const float* __restrict__ B1,
              const float* __restrict__ B3, bf16* __restrict__ Tout,
              int M, int N, int K) {
  __shared__ alignas(16) bf16 sA[128][32 + SA_PAD];
  __shared__ alignas(16) bf16 sB1[64][40];
  __shared__ alignas(16) bf16 sB3[64][40];
  int tid = threadIdx.x;
  int wave = tid >> 5, lane = tid & 31;
  int lr = lane & 15, lh = lane >> 4;
  int wm = (wave & 3) * 32, wn = (wave >> 2) * 32;
  int m0 = blockIdx.y * 128, n0 = blockIdx.x * 64;

  v8f ag[2][2], au[2][2];
#pragma unroll
  for (int i = 0; i < 2; i++)
#pragma unroll
    for (int j = 0; j < 2; j++) { ag[i][j] = vzero8f(); au[i][j] = vzero8f(); }

  int brow = tid >> 3, bcol = (tid & 7) * 8;

  for (int k0 = 0; k0 < K; k0 += 32) {
#if HAVE_TDM
    if (wave == 0) {
      tdm_load_tile_bf16(A + (size_t)m0 * K + k0,
                         (unsigned)(size_t)&sA[0][0],
                         M - m0, K - k0, 128, 32, K);
    }
#else
    {
      int ahalf = tid & 1;
      const bf16* ap = A + (size_t)(m0 + (tid >> 1)) * K + k0 + ahalf * 16;
      *(v8bf*)&sA[tid >> 1][ahalf * 16]     = *(const v8bf*)ap;
      *(v8bf*)&sA[tid >> 1][ahalf * 16 + 8] = *(const v8bf*)(ap + 8);
    }
#endif
    {
      const float* bp = B1 + (size_t)(k0 + brow) * N + n0 + bcol;
      const float* cp = B3 + (size_t)(k0 + brow) * N + n0 + bcol;
#pragma unroll
      for (int q = 0; q < 8; q++) {
        sB1[bcol + q][brow] = (bf16)bp[q];
        sB3[bcol + q][brow] = (bf16)cp[q];
      }
      if (k0 + 32 < K) {
        __builtin_prefetch(B1 + (size_t)(k0 + 32 + brow) * N + n0 + bcol, 0, 0);
        __builtin_prefetch(B3 + (size_t)(k0 + 32 + brow) * N + n0 + bcol, 0, 0);
      }
    }
#if HAVE_TDM
    if (wave == 0) __builtin_amdgcn_s_wait_tensorcnt(0);
#endif
    __syncthreads();
    v16bf af[2], b1f[2], b3f[2];
#pragma unroll
    for (int i = 0; i < 2; i++) {
      const bf16* ap = &sA[wm + i * 16 + lr][lh * 8];
      af[i] = cat8(*(const v8bf*)ap, *(const v8bf*)(ap + 16));
    }
#pragma unroll
    for (int j = 0; j < 2; j++) {
      const bf16* p1 = &sB1[wn + j * 16 + lr][lh * 8];
      const bf16* p3 = &sB3[wn + j * 16 + lr][lh * 8];
      b1f[j] = cat8(*(const v8bf*)p1, *(const v8bf*)(p1 + 16));
      b3f[j] = cat8(*(const v8bf*)p3, *(const v8bf*)(p3 + 16));
    }
#pragma unroll
    for (int i = 0; i < 2; i++)
#pragma unroll
      for (int j = 0; j < 2; j++) {
        ag[i][j] = wmma_bf16(af[i], b1f[j], ag[i][j]);
        au[i][j] = wmma_bf16(af[i], b3f[j], au[i][j]);
      }
    __syncthreads();
  }
#pragma unroll
  for (int i = 0; i < 2; i++)
#pragma unroll
    for (int j = 0; j < 2; j++)
#pragma unroll
      for (int r = 0; r < 8; r++) {
        int row = m0 + wm + i * 16 + r + 8 * lh;
        int col = n0 + wn + j * 16 + lr;
        float g = ag[i][j][r], u = au[i][j][r];
        /* silu(g)*u with v_rcp_f32 instead of the IEEE div expansion */
        float t = g * __builtin_amdgcn_rcpf(1.0f + __expf(-g)) * u;
        Tout[(size_t)row * N + col] = (bf16)t;
      }
}

/* ---------------- RoPE + layout to head-major bf16 ---------------- */
__global__ __launch_bounds__(256)
void k_rope(const float* __restrict__ src, bf16* __restrict__ dst, int do_rope) {
  int gid = blockIdx.x * 256 + threadIdx.x;
  int i = gid & 31;
  int h = (gid >> 5) & 7;
  int row = gid >> 8;
  int s = row & (S_SZ - 1), n = row >> 9;
  float x1 = src[(size_t)row * D_SZ + h * HD_SZ + i];
  float x2 = src[(size_t)row * D_SZ + h * HD_SZ + HALF_SZ + i];
  float o1 = x1, o2 = x2;
  if (do_rope) {
    float fr = __powf(10000.0f, -(float)i * (1.0f / HALF_SZ));
    float ang = (float)s * fr;
    float sn = sinf(ang), cs = cosf(ang);
    o1 = x1 * cs - x2 * sn;
    o2 = x1 * sn + x2 * cs;
  }
  size_t base = (((size_t)n * NH_SZ + h) * S_SZ + s) * HD_SZ;
  dst[base + i]           = (bf16)o1;
  dst[base + HALF_SZ + i] = (bf16)o2;
}

/* ---------------- flash attention ----------------
 * grid (160, 8): (n*8+h, qblock of 64). 128 threads = 4 waves x 16 q-rows. */
__global__ __launch_bounds__(128)
void k_attn(const bf16* __restrict__ qh, const bf16* __restrict__ kh,
            const bf16* __restrict__ vh, bf16* __restrict__ attn) {
  __shared__ alignas(16) bf16 sK[32][72];
  __shared__ alignas(16) bf16 sVt[64][40];
  __shared__ alignas(16) bf16 sP[4][16][40];
  int nh = blockIdx.x, qb = blockIdx.y;
  int n = nh >> 3, h = nh & 7;
  int tid = threadIdx.x, wave = tid >> 5, lane = tid & 31;
  int lr = lane & 15, lh = lane >> 4;
  size_t base = (size_t)nh * S_SZ * HD_SZ;
  int q0 = qb * 64 + wave * 16;
  const float scale = 0.125f;

  v16bf qf[2];
  {
    const bf16* qp = qh + base + (size_t)(q0 + lr) * HD_SZ;
#pragma unroll
    for (int ks = 0; ks < 2; ks++)
      qf[ks] = cat8(*(const v8bf*)(qp + ks * 32 + lh * 8),
                    *(const v8bf*)(qp + ks * 32 + lh * 8 + 16));
  }

  v8f accO[4];
#pragma unroll
  for (int f = 0; f < 4; f++) accO[f] = vzero8f();
  float mrow[8], srow[8];
#pragma unroll
  for (int r = 0; r < 8; r++) { mrow[r] = -1e30f; srow[r] = 0.0f; }

  int nkb = qb * 2 + 2;
  for (int kb = 0; kb < nkb; kb++) {
    __syncthreads();
    { /* stage K (row-major) and V (transposed) tiles, bf16 */
      int key = tid >> 2, seg = tid & 3;
      const bf16* kp = kh + base + (size_t)(kb * 32 + key) * HD_SZ + seg * 16;
      *(v8bf*)&sK[key][seg * 16]     = *(const v8bf*)kp;
      *(v8bf*)&sK[key][seg * 16 + 8] = *(const v8bf*)(kp + 8);
      const bf16* vp = vh + base + (size_t)(kb * 32 + key) * HD_SZ + seg * 16;
#pragma unroll
      for (int j = 0; j < 16; j++) sVt[seg * 16 + j][key] = vp[j];
    }
    __syncthreads();

    float sc[2][8];
#pragma unroll
    for (int t = 0; t < 2; t++) {
      v8f s_ = vzero8f();
#pragma unroll
      for (int ks = 0; ks < 2; ks++) {
        const bf16* bp = &sK[t * 16 + lr][ks * 32 + lh * 8];
        v16bf bfr = cat8(*(const v8bf*)bp, *(const v8bf*)(bp + 16));
        s_ = wmma_bf16(qf[ks], bfr, s_);
      }
#pragma unroll
      for (int r = 0; r < 8; r++) {
        int key = kb * 32 + t * 16 + lr;
        int qrow = q0 + r + 8 * lh;
        sc[t][r] = (key <= qrow) ? s_[r] * scale : -1e9f;
      }
    }

    float alpha8[8];
#pragma unroll
    for (int r = 0; r < 8; r++) {
      float m2 = fmaxf(sc[0][r], sc[1][r]);
#pragma unroll
      for (int off = 1; off < 16; off <<= 1)
        m2 = fmaxf(m2, __shfl_xor(m2, off, 32));
      float mnew = fmaxf(mrow[r], m2);
      float al = __expf(mrow[r] - mnew);
      float p0 = __expf(sc[0][r] - mnew);
      float p1 = __expf(sc[1][r] - mnew);
      float ls = p0 + p1;
#pragma unroll
      for (int off = 1; off < 16; off <<= 1)
        ls += __shfl_xor(ls, off, 32);
      srow[r] = srow[r] * al + ls;
      mrow[r] = mnew;
      alpha8[r] = al;
      sP[wave][r + 8 * lh][lr]      = (bf16)p0;
      sP[wave][r + 8 * lh][16 + lr] = (bf16)p1;
    }
#pragma unroll
    for (int f = 0; f < 4; f++)
#pragma unroll
      for (int r = 0; r < 8; r++) accO[f][r] *= alpha8[r];

    const bf16* pp = &sP[wave][lr][lh * 8];
    v16bf pf = cat8(*(const v8bf*)pp, *(const v8bf*)(pp + 16));
#pragma unroll
    for (int nt = 0; nt < 4; nt++) {
      const bf16* vp2 = &sVt[nt * 16 + lr][lh * 8];
      v16bf vfr = cat8(*(const v8bf*)vp2, *(const v8bf*)(vp2 + 16));
      accO[nt] = wmma_bf16(pf, vfr, accO[nt]);
    }
  }

  float rinv[8];
#pragma unroll
  for (int r = 0; r < 8; r++) rinv[r] = __builtin_amdgcn_rcpf(srow[r]);
#pragma unroll
  for (int nt = 0; nt < 4; nt++)
#pragma unroll
    for (int r = 0; r < 8; r++) {
      int q = q0 + r + 8 * lh;
      int d = h * HD_SZ + nt * 16 + lr;
      attn[(size_t)(n * S_SZ + q) * D_SZ + d] = (bf16)(accO[nt][r] * rinv[r]);
    }
}

/* ---------------- log-softmax over V + label pick (80 rows) ------------- */
__global__ __launch_bounds__(256)
void k_lse(const float* __restrict__ logits, const int* __restrict__ lids,
           float* __restrict__ rowlp) {
  __shared__ float red[256];
  int row = blockIdx.x;
  const float* lrp = logits + (size_t)row * V_SZ;
  int t = threadIdx.x;
  float mx = -1e30f;
  for (int i = t; i < V_SZ / 4; i += 256) {
    float4 v = ((const float4*)lrp)[i];
    mx = fmaxf(fmaxf(mx, v.x), fmaxf(v.y, fmaxf(v.z, v.w)));
  }
  red[t] = mx; __syncthreads();
#pragma unroll
  for (int o = 128; o > 0; o >>= 1) {
    if (t < o) red[t] = fmaxf(red[t], red[t + o]);
    __syncthreads();
  }
  mx = red[0]; __syncthreads();
  float sm = 0.0f;
  for (int i = t; i < V_SZ / 4; i += 256) {
    float4 v = ((const float4*)lrp)[i];
    sm += __expf(v.x - mx) + __expf(v.y - mx) + __expf(v.z - mx) + __expf(v.w - mx);
  }
  red[t] = sm; __syncthreads();
#pragma unroll
  for (int o = 128; o > 0; o >>= 1) {
    if (t < o) red[t] += red[t + o];
    __syncthreads();
  }
  if (t == 0) {
    float lse = mx + logf(red[0]);
    int nl = row >> 4;
    int j = row & 3;
    int tok = lids[nl * LBL_SZ + j];
    rowlp[row] = lrp[tok] - lse;
  }
}

__global__ void k_combine(const float* __restrict__ rowlp, float* __restrict__ out) {
  int idx = threadIdx.x;
  if (idx < B_SZ * NLAB_SZ) {
    int b = idx / NLAB_SZ, nl = idx % NLAB_SZ;
    int base = (nl * B_SZ + b) * LBL_SZ;
    out[idx] = rowlp[base] + rowlp[base + 1] + rowlp[base + 2] + rowlp[base + 3];
  }
}

/* ---------------- launcher ---------------- */
extern "C" void kernel_launch(void* const* d_in, const int* in_sizes, int n_in,
                              void* d_out, int out_size, void* d_ws, size_t ws_size,
                              hipStream_t stream) {
  (void)in_sizes; (void)n_in; (void)out_size; (void)ws_size;
  const int*   input_ids = (const int*)  d_in[0];
  const int*   label_ids = (const int*)  d_in[1];
  const float* emb       = (const float*)d_in[2];
  const float* wq        = (const float*)d_in[3];
  const float* wk        = (const float*)d_in[4];
  const float* wv        = (const float*)d_in[5];
  const float* wo        = (const float*)d_in[6];
  const float* w1        = (const float*)d_in[7];
  const float* w2        = (const float*)d_in[8];
  const float* w3        = (const float*)d_in[9];
  const float* attn_nw   = (const float*)d_in[10];
  const float* ffn_nw    = (const float*)d_in[11];
  const float* final_nw  = (const float*)d_in[12];
  const float* lm_head   = (const float*)d_in[13];
  float* out = (float*)d_out;

  char* wsb = (char*)d_ws;
  size_t off = 0;
  auto take = [&](size_t bytes) -> char* {
    char* p = wsb + off;
    off += (bytes + 255) & ~(size_t)255;
    return p;
  };
  float* h     = (float*)take((size_t)MROWS * D_SZ * 4);
  bf16*  xn    = (bf16*) take((size_t)MROWS * D_SZ * 2);
  float* qf    = (float*)take((size_t)MROWS * D_SZ * 4);
  float* kf    = (float*)take((size_t)MROWS * D_SZ * 4);
  float* vf    = (float*)take((size_t)MROWS * D_SZ * 4);
  bf16*  qhb   = (bf16*) take((size_t)MROWS * D_SZ * 2);
  bf16*  khb   = (bf16*) take((size_t)MROWS * D_SZ * 2);
  bf16*  vhb   = (bf16*) take((size_t)MROWS * D_SZ * 2);
  bf16*  attnb = (bf16*) take((size_t)MROWS * D_SZ * 2);
  bf16*  tb    = (bf16*) take((size_t)MROWS * HID_SZ * 2);
  bf16*  hsb   = (bf16*) take((size_t)80 * D_SZ * 2);
  float* rowlp = (float*)take(80 * 4);
  float* logits = qf;   /* alias: dead after layers; qf+kf region = 41.9MB >= 41MB */

  k_embed<<<MROWS, 128, 0, stream>>>(input_ids, label_ids, emb, h);

  dim3 g_dd(D_SZ / 64, MROWS / 128);
  dim3 g_dh(HID_SZ / 64, MROWS / 128);
  for (int l = 0; l < NL_SZ; l++) {
    const float* wql = wq + (size_t)l * D_SZ * D_SZ;
    const float* wkl = wk + (size_t)l * D_SZ * D_SZ;
    const float* wvl = wv + (size_t)l * D_SZ * D_SZ;
    const float* wol = wo + (size_t)l * D_SZ * D_SZ;
    const float* w1l = w1 + (size_t)l * D_SZ * HID_SZ;
    const float* w3l = w3 + (size_t)l * D_SZ * HID_SZ;
    const float* w2l = w2 + (size_t)l * HID_SZ * D_SZ;

    k_rmsnorm<<<MROWS, 256, 0, stream>>>(h, attn_nw + l * D_SZ, xn);
    k_gemm<0><<<g_dd, 256, 0, stream>>>(xn, wql, qf, MROWS, D_SZ, D_SZ);
    k_gemm<0><<<g_dd, 256, 0, stream>>>(xn, wkl, kf, MROWS, D_SZ, D_SZ);
    k_gemm<0><<<g_dd, 256, 0, stream>>>(xn, wvl, vf, MROWS, D_SZ, D_SZ);
    k_rope<<<MROWS, 256, 0, stream>>>(qf, qhb, 1);
    k_rope<<<MROWS, 256, 0, stream>>>(kf, khb, 1);
    k_rope<<<MROWS, 256, 0, stream>>>(vf, vhb, 0);
    k_attn<<<dim3(NSEQ * NH_SZ, S_SZ / 64), 128, 0, stream>>>(qhb, khb, vhb, attnb);
    k_gemm<1><<<g_dd, 256, 0, stream>>>(attnb, wol, h, MROWS, D_SZ, D_SZ);
    k_rmsnorm<<<MROWS, 256, 0, stream>>>(h, ffn_nw + l * D_SZ, xn);
    k_swiglu<<<g_dh, 256, 0, stream>>>(xn, w1l, w3l, tb, MROWS, HID_SZ, D_SZ);
    k_gemm<1><<<g_dd, 256, 0, stream>>>(tb, w2l, h, MROWS, D_SZ, HID_SZ);
  }

  k_rmsnorm_final<<<80, 256, 0, stream>>>(h, final_nw, hsb);
  k_gemm<0><<<dim3(V_SZ / 64, 1), 256, 0, stream>>>(hsb, lm_head, logits, 80, V_SZ, D_SZ);
  k_lse<<<80, 256, 0, stream>>>(logits, label_ids, rowlp);
  k_combine<<<1, 32, 0, stream>>>(rowlp, out);
}
